// wtGNNNorm_65549790871883
// MI455X (gfx1250) — matrix operand order repaired and checked by reading, not compiled, not run on hardware
//
#include <hip/hip_runtime.h>

#define NEG_SLOPE 0.2f
#define LN_EPS 1e-5f
#define ORD_NEG_INF 0x007FFFFFu   // ford(-inf)

typedef __attribute__((ext_vector_type(16))) __bf16 bf16x16;
typedef __attribute__((ext_vector_type(8)))  __bf16 bf16x8;
typedef __attribute__((ext_vector_type(8)))  float  f32x8;

union BFPack { bf16x16 v; bf16x8 h[2]; };

__device__ __forceinline__ unsigned short f2bf(float f) {
    unsigned u = __float_as_uint(f);
    unsigned r = 0x7fffu + ((u >> 16) & 1u);   // round-to-nearest-even
    return (unsigned short)((u + r) >> 16);
}
__device__ __forceinline__ unsigned ford(float f) {   // monotonic f32 -> u32
    unsigned u = __float_as_uint(f);
    return (u & 0x80000000u) ? ~u : (u | 0x80000000u);
}
__device__ __forceinline__ float funord(unsigned o) {
    unsigned u = (o & 0x80000000u) ? (o & 0x7fffffffu) : ~o;
    return __uint_as_float(u);
}

// ---- x f32 -> bf16, plus zero/init of per-layer scratch ------------------
__global__ void k_convert_x(const float* __restrict__ x, unsigned short* __restrict__ xbf,
                            float* __restrict__ agg, float* __restrict__ denom,
                            unsigned* __restrict__ mmax, int n_elems, int n4) {
    int i = blockIdx.x * 256 + threadIdx.x;
    if (i < n_elems) { xbf[i] = f2bf(x[i]); agg[i] = 0.f; }
    if (i < n4)      { denom[i] = 0.f; mmax[i] = ORD_NEG_INF; }
}

// ---- W (256x256 f32, row-major [k][n]) -> bf16 transposed [n][k] ----------
__global__ void k_convert_w(const float* __restrict__ Wl, const float* __restrict__ Wr,
                            unsigned short* __restrict__ WlT, unsigned short* __restrict__ WrT) {
    int i = blockIdx.x * 256 + threadIdx.x;     // 65536 elements
    int k = i >> 8, n = i & 255;
    WlT[n * 256 + k] = f2bf(Wl[k * 256 + n]);
    WrT[n * 256 + k] = f2bf(Wr[k * 256 + n]);
}

// ---- WMMA GEMM: out[N,256] = xbf[N,256] @ (wT^T)[256,256] -----------------
// One wave per 16x16 output tile; K-loop over 8 steps of K=32.
// A per-lane: b128 loads at (row, kb+hi*8) and +16  (matches 16-bit A layout)
// B per-lane: b128 loads at (colrow, kb+hi*8) and +16 (matches 16-bit B layout)
__global__ void k_gemm(const unsigned short* __restrict__ xbf,
                       const unsigned short* __restrict__ wT,
                       float* __restrict__ out, int tiles_m, int N) {
    int wave = (blockIdx.x * blockDim.x + threadIdx.x) >> 5;
    int lane = threadIdx.x & 31;
    int tm = wave >> 4;
    int tn = wave & 15;
    if (tm >= tiles_m) return;
    int hi = lane >> 4;
    int l  = lane & 15;
    int arow = tm * 16 + l;
    if (arow >= N) arow = N - 1;                // clamp ragged tail (reads only)
    const unsigned short* pa = xbf + (size_t)arow * 256 + hi * 8;
    const unsigned short* pb = wT  + (size_t)(tn * 16 + l) * 256 + hi * 8;
    f32x8 acc = {};
#pragma unroll
    for (int kb = 0; kb < 8; ++kb) {
        BFPack a, b;
        a.h[0] = *(const bf16x8*)(pa + kb * 32);
        a.h[1] = *(const bf16x8*)(pa + kb * 32 + 16);
        b.h[0] = *(const bf16x8*)(pb + kb * 32);
        b.h[1] = *(const bf16x8*)(pb + kb * 32 + 16);
        acc = __builtin_amdgcn_wmma_f32_16x16x32_bf16(
            false, a.v, false, b.v, (short)0, acc, false, false);
    }
    // C/D layout: VGPR r -> row r + hi*8, lanes 0-15 -> cols 0-15
    int rbase = tm * 16 + hi * 8;
#pragma unroll
    for (int r = 0; r < 8; ++r) {
        int row = rbase + r;
        if (row < N) out[(size_t)row * 256 + tn * 16 + l] = acc[r];
    }
}

// ---- per-edge attention scores + segment max ------------------------------
__global__ void k_score(const int* __restrict__ ei, const float* __restrict__ xl,
                        const float* __restrict__ xr, const float* __restrict__ att,
                        float* __restrict__ escore, unsigned* __restrict__ mmax,
                        int E, int Etot) {
    int wave = (blockIdx.x * blockDim.x + threadIdx.x) >> 5;
    int lane = threadIdx.x & 31;
    if (wave >= Etot) return;
    int s, d;
    if (wave < E) { s = ei[wave]; d = ei[E + wave]; }
    else          { s = wave - E; d = s; }
    const float4* pl = (const float4*)(xl + (size_t)s * 256 + lane * 8);
    const float4* pr = (const float4*)(xr + (size_t)d * 256 + lane * 8);
    const float4* pa = (const float4*)(att + lane * 8);
    float lv[8], rv[8], av[8];
    *(float4*)(lv)     = pl[0]; *(float4*)(lv + 4) = pl[1];
    *(float4*)(rv)     = pr[0]; *(float4*)(rv + 4) = pr[1];
    *(float4*)(av)     = pa[0]; *(float4*)(av + 4) = pa[1];
    float p = 0.f;
#pragma unroll
    for (int j = 0; j < 8; ++j) {
        float m = lv[j] + rv[j];
        p += (m > 0.f ? m : NEG_SLOPE * m) * av[j];
    }
    // reduce over the 8 lanes of each head (lanes 8h..8h+7)
    p += __shfl_xor(p, 1);
    p += __shfl_xor(p, 2);
    p += __shfl_xor(p, 4);
    if ((lane & 7) == 0) {
        int h = lane >> 3;
        escore[(size_t)wave * 4 + h] = p;
        atomicMax(&mmax[d * 4 + h], ford(p));
    }
}

// ---- w = exp(e - max), denom += w ----------------------------------------
__global__ void k_expsum(const int* __restrict__ ei, float* __restrict__ escore,
                         const unsigned* __restrict__ mmax, float* __restrict__ denom,
                         int E, int Etot) {
    int i = blockIdx.x * 256 + threadIdx.x;
    if (i >= Etot * 4) return;
    int e = i >> 2, h = i & 3;
    int d = (e < E) ? ei[E + e] : (e - E);
    float m = funord(mmax[d * 4 + h]);
    float w = expf(escore[i] - m);
    escore[i] = w;
    atomicAdd(&denom[d * 4 + h], w);
}

// ---- out[d] += alpha * xl[s] ---------------------------------------------
__global__ void k_aggregate(const int* __restrict__ ei, const float* __restrict__ xl,
                            const float* __restrict__ escore, const float* __restrict__ denom,
                            float* __restrict__ agg, int E, int Etot) {
    int wave = (blockIdx.x * blockDim.x + threadIdx.x) >> 5;
    int lane = threadIdx.x & 31;
    if (wave >= Etot) return;
    int s, d;
    if (wave < E) { s = ei[wave]; d = ei[E + wave]; }
    else          { s = wave - E; d = s; }
    int h = lane >> 3;
    float alpha = escore[(size_t)wave * 4 + h] / denom[d * 4 + h];
    const float4* pl = (const float4*)(xl + (size_t)s * 256 + lane * 8);
    float lv[8];
    *(float4*)(lv) = pl[0]; *(float4*)(lv + 4) = pl[1];
    float* po = agg + (size_t)d * 256 + lane * 8;
#pragma unroll
    for (int j = 0; j < 8; ++j) atomicAdd(po + j, alpha * lv[j]);
}

// ---- bias + LayerNorm + ReLU (one block of 256 per node) ------------------
__global__ void k_finalize(const float* __restrict__ agg, const float* __restrict__ bias,
                           const float* __restrict__ g, const float* __restrict__ be,
                           float* __restrict__ outp) {
    int n = blockIdx.x, t = threadIdx.x;
    float v = agg[(size_t)n * 256 + t] + bias[t];
    float s = v, q = v * v;
#pragma unroll
    for (int o = 16; o >= 1; o >>= 1) { s += __shfl_xor(s, o); q += __shfl_xor(q, o); }
    __shared__ float ss[8], sq[8];
    int wid = t >> 5;
    if ((t & 31) == 0) { ss[wid] = s; sq[wid] = q; }
    __syncthreads();
    float tot = 0.f, totq = 0.f;
#pragma unroll
    for (int i = 0; i < 8; ++i) { tot += ss[i]; totq += sq[i]; }
    float mu  = tot  * (1.f / 256.f);
    float var = totq * (1.f / 256.f) - mu * mu;
    float y = (v - mu) * rsqrtf(var + LN_EPS) * g[t] + be[t];
    outp[(size_t)n * 256 + t] = fmaxf(y, 0.f);
}

extern "C" void kernel_launch(void* const* d_in, const int* in_sizes, int n_in,
                              void* d_out, int out_size, void* d_ws, size_t ws_size,
                              hipStream_t stream) {
    (void)n_in; (void)out_size; (void)ws_size;
    const float* x0 = (const float*)d_in[0];
    const int*   ei = (const int*)d_in[1];
    const int N    = in_sizes[0] / 256;
    const int E    = in_sizes[1] / 2;
    const int Etot = E + N;

    char* ws = (char*)d_ws;
    size_t off = 0;
    auto alloc = [&](size_t bytes) -> void* {
        void* p = ws + off;
        off = (off + bytes + 255) & ~(size_t)255;
        return p;
    };
    unsigned short* xbf  = (unsigned short*)alloc((size_t)N * 256 * 2);
    unsigned short* wlT  = (unsigned short*)alloc((size_t)256 * 256 * 2);
    unsigned short* wrT  = (unsigned short*)alloc((size_t)256 * 256 * 2);
    float*    xl     = (float*)alloc((size_t)N * 256 * 4);
    float*    xr     = (float*)alloc((size_t)N * 256 * 4);
    float*    escore = (float*)alloc((size_t)Etot * 4 * 4);
    unsigned* mmax   = (unsigned*)alloc((size_t)N * 4 * 4);
    float*    denom  = (float*)alloc((size_t)N * 4 * 4);
    float*    agg    = (float*)alloc((size_t)N * 256 * 4);
    float*    hbuf   = (float*)alloc((size_t)N * 256 * 4);

    const int nvec        = N * 256;
    const int tiles_m     = (N + 15) / 16;
    const int gemm_waves  = tiles_m * 16;
    const int gemm_blocks = (gemm_waves + 7) / 8;       // 8 waves per 256-thread block
    const int conv_blocks = (nvec + 255) / 256;
    const int edge_blocks = (Etot * 32 + 255) / 256;    // one wave per edge
    const int exp_blocks  = (Etot * 4 + 255) / 256;

    const float* hcur = x0;
    for (int layer = 0; layer < 4; ++layer) {
        const float* Wl   = (const float*)d_in[2 + layer * 6 + 0];
        const float* Wr   = (const float*)d_in[2 + layer * 6 + 1];
        const float* att  = (const float*)d_in[2 + layer * 6 + 2];
        const float* bias = (const float*)d_in[2 + layer * 6 + 3];
        const float* gam  = (const float*)d_in[2 + layer * 6 + 4];
        const float* bet  = (const float*)d_in[2 + layer * 6 + 5];
        float* outp = (layer == 3) ? (float*)d_out : hbuf;

        k_convert_x<<<conv_blocks, 256, 0, stream>>>(hcur, xbf, agg, denom, mmax, nvec, N * 4);
        k_convert_w<<<256, 256, 0, stream>>>(Wl, Wr, wlT, wrT);
        k_gemm<<<gemm_blocks, 256, 0, stream>>>(xbf, wlT, xl, tiles_m, N);
        k_gemm<<<gemm_blocks, 256, 0, stream>>>(xbf, wrT, xr, tiles_m, N);
        k_score<<<edge_blocks, 256, 0, stream>>>(ei, xl, xr, att, escore, mmax, E, Etot);
        k_expsum<<<exp_blocks, 256, 0, stream>>>(ei, escore, mmax, denom, E, Etot);
        k_aggregate<<<edge_blocks, 256, 0, stream>>>(ei, xl, escore, denom, agg, E, Etot);
        k_finalize<<<N, 256, 0, stream>>>(agg, bias, gam, bet, outp);
        hcur = hbuf;
    }
}
